// GATlayer_10617159155774
// MI455X (gfx1250) — matrix operand order repaired and checked by reading, not compiled, not run on hardware
//
#include <hip/hip_runtime.h>
#include <hip/hip_bf16.h>
#include <math.h>

// ---------------------------------------------------------------------------
// GAT network for MI455X (gfx1250, wave32).  GEMMs use v_wmma_f32_16x16x32_bf16
// (fp32 accumulate); softmax / BN / scatter stay fp32.  Bandwidth-bound on the
// edge scatter; all hot buffers fit in the 192MB L2.
// B tiles are staged into LDS in *fragment order* so each WMMA B operand is
// fetched with 2x ds_load_b128; staging loads are unpredicated on the fast
// (full-K-tile) path so the 8 strided global loads pipeline under one wait.
// ---------------------------------------------------------------------------

typedef __attribute__((ext_vector_type(16))) __bf16 v16bf;
typedef __attribute__((ext_vector_type(8)))  __bf16 v8bf;
typedef __attribute__((ext_vector_type(8)))  float  v8f;

#define NHEADS 8
#define EPSBN  1e-5f
#define OUTC   456   // 256 pooled + 200 rdkit

// ------------------------------ GEMM (WMMA) --------------------------------
// C[M,Nc] = A[M,K] * B[K,Nc] (+bias).  Block = 256 threads = 8 waves.
// Each wave computes a 16 (M) x 64 (N) strip: 4 f32 accumulators, K-step 32.
// A fragment layout (ISA 7.12.2, 16-bit A 16x32): lane&15 = row,
//   lanes 0-15 hold K octets {0..7,16..23}, lanes 16-31 hold {8..15,24..31}.
// B fragment: lane&15 = col, lanes 0-15 hold K {0..7,16..23}, lanes 16-31 {8..15,24..31} (as A).
// LDS layout: Bsh[(j*32 + lane)*16 + e] -> per-lane fragment = 32 contiguous B.
__global__ __launch_bounds__(256)
void gemm_bf16_wmma(const float* __restrict__ A, const float* __restrict__ B,
                    const float* __restrict__ bias, float* __restrict__ C,
                    int M, int K, int Nc) {
  __shared__ __align__(16) __bf16 Bsh[4 * 32 * 16];   // 4 j-subtiles x 32 lanes x 16 bf16

  const int tid  = threadIdx.x;
  const int lane = tid & 31;
  const int wid  = tid >> 5;
  const int rowBase = blockIdx.x * 128 + wid * 16;
  const int colBase = blockIdx.y * 64;
  const int arow = rowBase + (lane & 15);
  const int kb   = (lane >> 4) << 3;     // A: 0 or 8

  // B staging coordinates (per thread, loop-invariant)
  const int cc    = tid & 63;            // column within 64-wide tile
  const int ks    = tid >> 6;            // K octet 0..3
  const int sj    = cc >> 4;             // j-subtile
  const int slane = (cc & 15) + ((ks & 1) << 4);
  const int ebase = (ks >> 1) * 8;
  const bool colOK = (colBase + cc) < Nc;            // uniform given Nc%64==0
  __bf16* sdst = &Bsh[(((sj * 32 + slane) << 4) + ebase)];
  const float* bcol = B + (size_t)ks * 8 * Nc + colBase + cc;

  const bool fastA = ((K & 3) == 0);

  v8f z = {};
  v8f acc[4];
  acc[0] = z; acc[1] = z; acc[2] = z; acc[3] = z;

  for (int k0 = 0; k0 < K; k0 += 32) {
    // ---- stage B tile [32 x 64] -> LDS in fragment order ------------------
    {
      const float* bp = bcol + (size_t)k0 * Nc;
      v8bf t;
      if ((k0 + 32) <= K && colOK) {
        // fast path: no per-element predicates -> 8 pipelined global loads
        #pragma unroll
        for (int i = 0; i < 8; ++i) t[i] = (__bf16)bp[(size_t)i * Nc];
      } else {
        #pragma unroll
        for (int i = 0; i < 8; ++i) {
          int gk = k0 + ks * 8 + i;
          float v = (gk < K && colOK) ? bp[(size_t)i * Nc] : 0.0f;
          t[i] = (__bf16)v;
        }
      }
      *(v8bf*)sdst = t;                  // one ds_store_b128
    }
    __syncthreads();

    // ---- A fragment straight from global (fp32 -> bf16) -------------------
    v16bf af;
    const float* ap = A + (size_t)arow * K;
    if (fastA && arow < M && (k0 + 32) <= K) {
      const float4* p = (const float4*)(ap + k0 + kb);        // 16B aligned
      const float4* q = (const float4*)(ap + k0 + 16 + kb);
      float4 a0 = p[0], a1 = p[1], b0 = q[0], b1 = q[1];
      af[0]  = (__bf16)a0.x; af[1]  = (__bf16)a0.y; af[2]  = (__bf16)a0.z; af[3]  = (__bf16)a0.w;
      af[4]  = (__bf16)a1.x; af[5]  = (__bf16)a1.y; af[6]  = (__bf16)a1.z; af[7]  = (__bf16)a1.w;
      af[8]  = (__bf16)b0.x; af[9]  = (__bf16)b0.y; af[10] = (__bf16)b0.z; af[11] = (__bf16)b0.w;
      af[12] = (__bf16)b1.x; af[13] = (__bf16)b1.y; af[14] = (__bf16)b1.z; af[15] = (__bf16)b1.w;
    } else {
      #pragma unroll
      for (int e = 0; e < 16; ++e) {
        int k = (e < 8) ? (kb + e) : (16 + kb + (e - 8));
        float v = (arow < M && (k0 + k) < K) ? ap[k0 + k] : 0.0f;
        af[e] = (__bf16)v;
      }
    }

    // ---- 4 WMMA per K step ------------------------------------------------
    const v16bf* bsp = (const v16bf*)Bsh;
    #pragma unroll
    for (int j = 0; j < 4; ++j) {
      v16bf bf = bsp[j * 32 + lane];     // 2x ds_load_b128
      acc[j] = __builtin_amdgcn_wmma_f32_16x16x32_bf16(
          false, af, false, bf, (short)0, acc[j], false, false);
    }
    __syncthreads();
  }

  // ---- epilogue: C/D layout -> lane&15 = col, vgpr v = row (lane>=16:+8) --
  const int ncol = lane & 15;
  const int moff = (lane >> 4) * 8;
  #pragma unroll
  for (int j = 0; j < 4; ++j) {
    int col = colBase + j * 16 + ncol;
    if (col >= Nc) continue;
    float bv = bias ? bias[col] : 0.0f;
    #pragma unroll
    for (int vv = 0; vv < 8; ++vv) {
      int row = rowBase + moff + vv;
      if (row < M) C[(size_t)row * Nc + col] = acc[j][vv] + bv;
    }
  }
}

// ------------------------------ helpers ------------------------------------
__global__ void fill_kernel(float* p, float v, int n) {
  int i = blockIdx.x * blockDim.x + threadIdx.x;
  if (i < n) p[i] = v;
}

__device__ inline void edge_ends(const int* __restrict__ src,
                                 const int* __restrict__ dst,
                                 int e, int E, int& s, int& d) {
  if (e < E) { s = src[e]; d = dst[e]; }
  else       { s = e - E;  d = e - E;  }      // virtual self-loops
}

__device__ inline float atomicMaxF(float* addr, float val) {
  unsigned int* ua = (unsigned int*)addr;
  unsigned int old = *ua;
  while (__uint_as_float(old) < val) {
    unsigned int assumed = old;
    old = atomicCAS(ua, assumed, __float_as_uint(val));
    if (old == assumed) break;
  }
  return __uint_as_float(old);
}

// a_src[n,h] = <hW[n,h,:], att_src[h,:]>, same for dst
__global__ void node_att_kernel(const float* __restrict__ hw,
                                const float* __restrict__ as,
                                const float* __restrict__ ad,
                                float* __restrict__ asrc, float* __restrict__ adst,
                                int N, int C) {
  int i = blockIdx.x * blockDim.x + threadIdx.x;
  if (i >= N * NHEADS) return;
  int n = i / NHEADS, h = i % NHEADS;
  const float* row = hw + (size_t)n * (NHEADS * C) + h * C;
  float s1 = 0.f, s2 = 0.f;
  for (int c = 0; c < C; ++c) { s1 += row[c] * as[h * C + c]; s2 += row[c] * ad[h * C + c]; }
  asrc[i] = s1; adst[i] = s2;
}

__global__ void edge_max_kernel(const int* src, const int* dst,
                                const float* asrc, const float* adst,
                                float* m, int E, int ET) {
  int i = blockIdx.x * blockDim.x + threadIdx.x;
  if (i >= ET * NHEADS) return;
  int e = i / NHEADS, h = i % NHEADS, s, d;
  edge_ends(src, dst, e, E, s, d);
  float v = asrc[s * NHEADS + h] + adst[d * NHEADS + h];
  v = v > 0.f ? v : 0.2f * v;                           // leaky_relu 0.2
  atomicMaxF(&m[d * NHEADS + h], v);
}

__global__ void edge_expsum_kernel(const int* src, const int* dst,
                                   const float* asrc, const float* adst,
                                   const float* m, float* exbuf, float* sbuf,
                                   int E, int ET) {
  int i = blockIdx.x * blockDim.x + threadIdx.x;
  if (i >= ET * NHEADS) return;
  int e = i / NHEADS, h = i % NHEADS, s, d;
  edge_ends(src, dst, e, E, s, d);
  float v = asrc[s * NHEADS + h] + adst[d * NHEADS + h];
  v = v > 0.f ? v : 0.2f * v;
  float ex = __expf(v - m[d * NHEADS + h]);
  exbuf[i] = ex;
  atomicAdd(&sbuf[d * NHEADS + h], ex);
}

// gacc[dst, c] += (ex/s[dst]) * hW[src, c]  — one block per edge
__global__ void edge_scatter_kernel(const int* src, const int* dst,
                                    const float* __restrict__ exbuf,
                                    const float* __restrict__ sbuf,
                                    const float* __restrict__ hw,
                                    float* __restrict__ gacc,
                                    int E, int C, int HC) {
  int e = blockIdx.x, s, d;
  edge_ends(src, dst, e, E, s, d);
  for (int c = threadIdx.x; c < HC; c += blockDim.x) {
    int h = c / C;
    float alpha = exbuf[(size_t)e * NHEADS + h] / sbuf[d * NHEADS + h];
    atomicAdd(&gacc[(size_t)d * HC + c], alpha * hw[(size_t)s * HC + c]);
  }
}

__global__ void bias_relu_kernel(float* x, const float* bias, int n, int HC) {
  int i = blockIdx.x * blockDim.x + threadIdx.x;
  if (i >= n) return;
  float v = x[i] + bias[i % HC];
  x[i] = v > 0.f ? v : 0.f;
}

__global__ void col_stats_kernel(const float* __restrict__ x,
                                 float* sum, float* sumsq, int N, int HC) {
  int col = blockIdx.x * blockDim.x + threadIdx.x;
  if (col >= HC) return;
  float s1 = 0.f, s2 = 0.f;
  for (int r = blockIdx.y; r < N; r += gridDim.y) {
    float v = x[(size_t)r * HC + col];
    s1 += v; s2 += v * v;
  }
  atomicAdd(&sum[col], s1);
  atomicAdd(&sumsq[col], s2);
}

// out = bn(x)  (accumulate==0)  or  out += bn(x)  (accumulate==1)
__global__ void bn_apply_kernel(const float* __restrict__ x,
                                const float* sum, const float* sumsq,
                                const float* gamma, const float* beta,
                                float* out, int N, int HC, int accumulate) {
  int i = blockIdx.x * blockDim.x + threadIdx.x;
  if (i >= N * HC) return;
  int c = i % HC;
  float inv = 1.0f / (float)N;
  float mu  = sum[c] * inv;
  float var = sumsq[c] * inv - mu * mu;
  float y = (x[i] - mu) * rsqrtf(var + EPSBN) * gamma[c] + beta[c];
  out[i] = accumulate ? (out[i] + y) : y;
}

// init output: pooled region = 0, rdkit features copied
__global__ void out_init_kernel(float* out, const float* gfr, int G) {
  int i = blockIdx.x * blockDim.x + threadIdx.x;
  if (i >= G * OUTC) return;
  int g = i / OUTC, c = i % OUTC;
  out[i] = (c < 256) ? 0.0f : gfr[g * 200 + (c - 256)];
}

__global__ void pool_kernel(const float* __restrict__ g4,
                            const int* __restrict__ batch,
                            float* out, int N) {
  int i = blockIdx.x * blockDim.x + threadIdx.x;
  if (i >= N * 256) return;
  int n = i >> 8, c = i & 255;
  atomicAdd(&out[(size_t)batch[n] * OUTC + c], g4[(size_t)n * 256 + c]);
}

// ------------------------------ host orchestration -------------------------
extern "C" void kernel_launch(void* const* d_in, const int* in_sizes, int n_in,
                              void* d_out, int out_size, void* d_ws, size_t ws_size,
                              hipStream_t stream) {
  const float* x     = (const float*)d_in[0];
  const int*   ei    = (const int*)  d_in[1];
  const int*   batch = (const int*)  d_in[3];
  const float* gfr   = (const float*)d_in[4];

  const int E  = in_sizes[1] / 2;     // 160000
  const int N  = in_sizes[3];         // 20000
  const int ET = E + N;               // + self loops
  const int G  = in_sizes[4] / 200;   // 128
  const int* src = ei;
  const int* dst = ei + E;

  const float* P[64];
  for (int i = 0; i < n_in && i < 64; ++i) P[i] = (const float*)d_in[i];
  // param layout (setup_inputs dict order, depth-first):
  // 5..10: conv1{W,as,ad,bias} bn1{g,b}; 11..16 conv2/bn2; 17..22 conv3/bn3;
  // 23..28 conv4/bn4; 29..32 res1{W,b,g,beta}; 33..36 res2; 37..40 res3.

  // ---- workspace carve (floats) ----
  float* ws   = (float*)d_ws;
  float* h    = ws;  ws += (size_t)N * 512;   // current features
  float* hw   = ws;  ws += (size_t)N * 512;   // h@W (gat), later reused for res
  float* gacc = ws;  ws += (size_t)N * 512;   // attention aggregation
  float* asrc = ws;  ws += (size_t)N * NHEADS;
  float* adst = ws;  ws += (size_t)N * NHEADS;
  float* mbuf = ws;  ws += (size_t)N * NHEADS;
  float* sbuf = ws;  ws += (size_t)N * NHEADS;
  float* exb  = ws;  ws += (size_t)ET * NHEADS;
  float* st   = ws;  ws += 4 * 512;           // sum_g, sq_g, sum_r, sq_r

  const int TB = 256;
  auto blks = [](long n, int b) { return (unsigned)((n + b - 1) / b); };

  // GAT branch for one layer: hin[N,K] -> gacc[N, 8*C] = relu(GATConv + bias)
  auto run_gat = [&](const float* hin, int K, int C, int base) {
    const int HC = NHEADS * C;
    dim3 gg(blks(N, 128), blks(HC, 64));
    gemm_bf16_wmma<<<gg, TB, 0, stream>>>(hin, P[base + 0], nullptr, hw, N, K, HC);
    node_att_kernel<<<blks((long)N * NHEADS, TB), TB, 0, stream>>>(
        hw, P[base + 1], P[base + 2], asrc, adst, N, C);
    fill_kernel<<<blks((long)N * NHEADS, TB), TB, 0, stream>>>(mbuf, -1e30f, N * NHEADS);
    fill_kernel<<<blks((long)N * NHEADS, TB), TB, 0, stream>>>(sbuf, 0.0f, N * NHEADS);
    fill_kernel<<<blks((long)N * HC, TB), TB, 0, stream>>>(gacc, 0.0f, N * HC);
    edge_max_kernel<<<blks((long)ET * NHEADS, TB), TB, 0, stream>>>(
        src, dst, asrc, adst, mbuf, E, ET);
    edge_expsum_kernel<<<blks((long)ET * NHEADS, TB), TB, 0, stream>>>(
        src, dst, asrc, adst, mbuf, exb, sbuf, E, ET);
    edge_scatter_kernel<<<ET, TB, 0, stream>>>(src, dst, exb, sbuf, hw, gacc, E, C, HC);
    bias_relu_kernel<<<blks((long)N * HC, TB), TB, 0, stream>>>(gacc, P[base + 3], N * HC, HC);
  };

  // ---- layers 1..3 (C=64, HC=512) ----
  const int convBase[4] = {5, 11, 17, 23};
  const int resBase[3]  = {29, 33, 37};
  const float* hin = x;
  int K = 38;
  for (int li = 0; li < 3; ++li) {
    const int HC = 512, cb = convBase[li], rb = resBase[li];
    run_gat(hin, K, 64, cb);
    // BN stats of GAT branch
    fill_kernel<<<blks(1024, TB), TB, 0, stream>>>(st, 0.0f, 1024);
    col_stats_kernel<<<dim3(blks(HC, TB), 128), TB, 0, stream>>>(gacc, st, st + 512, N, HC);
    // residual branch: hin @ W + b -> hw (hw's GAT contents no longer needed)
    dim3 gr(blks(N, 128), blks(HC, 64));
    gemm_bf16_wmma<<<gr, TB, 0, stream>>>(hin, P[rb + 0], P[rb + 1], hw, N, K, HC);
    fill_kernel<<<blks(1024, TB), TB, 0, stream>>>(st + 1024, 0.0f, 1024);
    col_stats_kernel<<<dim3(blks(HC, TB), 128), TB, 0, stream>>>(hw, st + 1024, st + 1536, N, HC);
    // h = bn(gacc; bn_g, bn_b) + bn(hw; res_g, res_beta)
    bn_apply_kernel<<<blks((long)N * HC, TB), TB, 0, stream>>>(
        gacc, st, st + 512, P[cb + 4], P[cb + 5], h, N, HC, 0);
    bn_apply_kernel<<<blks((long)N * HC, TB), TB, 0, stream>>>(
        hw, st + 1024, st + 1536, P[rb + 2], P[rb + 3], h, N, HC, 1);
    hin = h; K = 512;
  }

  // ---- layer 4 (C=32, HC=256) + pool + concat ----
  {
    const int HC = 256, cb = convBase[3];
    run_gat(h, 512, 32, cb);
    fill_kernel<<<blks(1024, TB), TB, 0, stream>>>(st, 0.0f, 1024);
    col_stats_kernel<<<dim3(blks(HC, TB), 128), TB, 0, stream>>>(gacc, st, st + 512, N, HC);
    bn_apply_kernel<<<blks((long)N * HC, TB), TB, 0, stream>>>(
        gacc, st, st + 512, P[cb + 4], P[cb + 5], gacc, N, HC, 0);
    out_init_kernel<<<blks((long)G * OUTC, TB), TB, 0, stream>>>((float*)d_out, gfr, G);
    pool_kernel<<<blks((long)N * 256, TB), TB, 0, stream>>>(gacc, batch, (float*)d_out, N);
  }
  (void)out_size; (void)ws_size;
}
